// CBAM_84799834292534
// MI455X (gfx1250) — matrix operand order repaired and checked by reading, not compiled, not run on hardware
//
#include <hip/hip_runtime.h>
#include <math.h>

// ---------------------------------------------------------------------------
// Types for CDNA5 WMMA (wave32, 16x16x32 bf16 -> f32) and TDM descriptors
// ---------------------------------------------------------------------------
typedef __attribute__((ext_vector_type(4)))  unsigned int v4u;
typedef __attribute__((ext_vector_type(8)))  unsigned int v8u;
typedef __attribute__((ext_vector_type(16))) __bf16       v16bf;
typedef __attribute__((ext_vector_type(8)))  float        v8f;
typedef __attribute__((ext_vector_type(4)))  unsigned int u32x4;
typedef __attribute__((ext_vector_type(8)))  int          i32x8;
typedef __attribute__((ext_vector_type(4)))  int          i32x4;

union A16 { v4u q[2]; v16bf b; };
union B16 { v8u u;    v16bf b; };

#if __has_builtin(__builtin_amdgcn_tensor_load_to_lds) && __has_builtin(__builtin_amdgcn_s_wait_tensorcnt)
#define HAVE_TDM 1
#else
#define HAVE_TDM 0
#endif

__device__ __forceinline__ unsigned short f2bf(float f) {
  unsigned int u = __float_as_uint(f);
  u += 0x7FFFu + ((u >> 16) & 1u);          // round-to-nearest-even
  return (unsigned short)(u >> 16);
}
__device__ __forceinline__ float bf2f(unsigned short h) {
  return __uint_as_float(((unsigned int)h) << 16);
}
__device__ __forceinline__ float sigm(float x) { return 1.0f / (1.0f + expf(-x)); }

// ---------------------------------------------------------------------------
// TDM: 1-D flat copy global -> LDS.  D# per CDNA5 ISA:
// group0 = {count=1 | lds_addr | global_addr | type=2}, group1 carries
// data_size=8B, tensor/tile dim0 = nbytes/8, stride = dim0.
// ---------------------------------------------------------------------------
__device__ __forceinline__ void tdm_load_1d(unsigned lds_addr,
                                            unsigned long long gaddr,
                                            unsigned nbytes)
{
#if HAVE_TDM
  const unsigned n8 = nbytes >> 3;            // 8-byte elements (< 2^16 here)
  u32x4 g0;
  g0.x = 1u;                                           // count=1, user mode
  g0.y = lds_addr;                                     // lds_addr [63:32]
  g0.z = (unsigned)(gaddr & 0xFFFFFFFFull);            // global_addr[31:0]
  g0.w = (unsigned)((gaddr >> 32) & 0x01FFFFFFull)     // global_addr[56:32]
         | 0x80000000u;                                // type=2 ("image")
  i32x8 g1;
  g1[0] = (int)(3u << 16);                             // data_size = 8B
  g1[1] = (int)((n8 & 0xFFFFu) << 16);                 // tensor_dim0[15:0]
  g1[2] = (int)(((n8 >> 16) & 0xFFFFu) | (1u << 16));  // dim0 hi | tensor_dim1=1
  g1[3] = (int)((n8 & 0xFFFFu) << 16);                 // tile_dim0 = n8
  g1[4] = 1;                                           // tile_dim1 = 1
  g1[5] = (int)n8;                                     // tensor_dim0_stride lo
  g1[6] = (int)((n8 & 0xFFFFu) << 16);                 // tensor_dim1_stride lo
  g1[7] = (int)(n8 >> 16);
  i32x4 gz = {0, 0, 0, 0};
#if defined(__clang_major__) && (__clang_major__ >= 23)
  i32x8 gz8 = {0, 0, 0, 0, 0, 0, 0, 0};
  __builtin_amdgcn_tensor_load_to_lds(g0, g1, gz, gz, gz8, 0);
#else
  __builtin_amdgcn_tensor_load_to_lds(g0, g1, gz, gz, 0);
#endif
#else
  (void)lds_addr; (void)gaddr; (void)nbytes;
#endif
}

// ---------------------------------------------------------------------------
// Implicit-GEMM convolution on v_wmma_f32_16x16x32_bf16.
// One wave = one 16(M=pixels) x 16(N=Cout) tile.  Weights are pre-swizzled
// into per-lane fragment layout [ntile][tap][ci_chunk][lane][8 dwords] and
// staged per-block into LDS via the Tensor Data Mover (all 8 waves of a
// block share blockIdx.y's n-tile, so the block's weights are contiguous).
// A fragments are two aligned global_load_b128 per chunk (NHWC bf16).
// ---------------------------------------------------------------------------
template <int KSIZE, int REFLECT, int CIN>
__global__ void __launch_bounds__(256)
conv_wmma_kernel(const unsigned short* __restrict__ act,
                 const unsigned int* __restrict__ wfrag,
                 const float* __restrict__ bias,
                 float* __restrict__ out,
                 int Bn, int Hin, int Win, int Cout, int Hout, int Wout)
{
  constexpr int NC   = CIN / 32;          // K chunks per tap
  constexpr int TAPS = KSIZE * KSIZE;
  const int lane = threadIdx.x & 31;
  const int wave = threadIdx.x >> 5;
  const int half = lane >> 4;
  const int mrow = lane & 15;
  const int ncol = lane & 15;

  // ---- stage this block's weight fragments into LDS (TDM, once) ----------
  extern __shared__ unsigned char smem_raw[];
  const unsigned int* wblk = wfrag + (size_t)blockIdx.y * (TAPS * NC * 256);
#if HAVE_TDM
  if (wave == 0)
    tdm_load_1d((unsigned)(uintptr_t)smem_raw,
                (unsigned long long)(uintptr_t)wblk,
                (unsigned)(TAPS * NC * 1024));
  __builtin_amdgcn_s_wait_tensorcnt(0);
#else
  for (int i = threadIdx.x; i < TAPS * NC * 256; i += 256)
    ((unsigned int*)smem_raw)[i] = wblk[i];
#endif
  __syncthreads();

  const int Mtot = Bn * Hout * Wout;      // divisible by 16
  const int m0   = (blockIdx.x * 8 + wave) * 16;
  if (m0 >= Mtot) return;                 // wave-uniform; barrier already done
  const int n0 = blockIdx.y * 16;

  // decode this lane's output pixel (row m0+mrow of the A matrix)
  int m  = m0 + mrow;
  int ox = m % Wout;
  int t  = m / Wout;
  int oy = t % Hout;
  int b  = t / Hout;

  v8f acc = {};
  const v4u* actq = (const v4u*)act;      // 8 bf16 elements per v4u
  const v8u* lw   = (const v8u*)smem_raw; // LDS weight fragments
  const v4u vzero = {0u, 0u, 0u, 0u};

  for (int ky = 0; ky < KSIZE; ++ky) {
    int iy; bool vy;
    if (REFLECT) {
      int py = oy + ky - 1;               // coord in reflect-padded image
      vy = (py >= 0) && (py < Hin + 2);   // conv's own zero-pad
      iy = py - 1;
      iy = iy < 0 ? 1 : (iy >= Hin ? Hin - 2 : iy);
    } else {
      iy = oy + ky - (KSIZE / 2);
      vy = (iy >= 0) && (iy < Hin);
      iy = iy < 0 ? 0 : (iy >= Hin ? Hin - 1 : iy);   // keep address in range
    }
#pragma unroll
    for (int kx = 0; kx < KSIZE; ++kx) {
      int ix; bool vx;
      if (REFLECT) {
        int px = ox + kx - 1;
        vx = (px >= 0) && (px < Win + 2);
        ix = px - 1;
        ix = ix < 0 ? 1 : (ix >= Win ? Win - 2 : ix);
      } else {
        ix = ox + kx - (KSIZE / 2);
        vx = (ix >= 0) && (ix < Win);
        ix = ix < 0 ? 0 : (ix >= Win ? Win - 1 : ix);
      }
      const bool valid = vy && vx;
      const int arow = ((b * Hin + iy) * Win + ix) * CIN;   // bf16 elements
      const int tap  = ky * KSIZE + kx;
      __builtin_prefetch((const void*)(act + arow), 0, 3);

#pragma unroll
      for (int cc = 0; cc < NC; ++cc) {
        A16 aa;
        aa.q[0] = valid ? actq[((arow + cc * 32)      >> 3) + 2 * half] : vzero;
        aa.q[1] = valid ? actq[((arow + cc * 32 + 8)  >> 3) + 2 * half] : vzero;
        B16 bb;
        bb.u = lw[(tap * NC + cc) * 32 + lane];             // ds_load_b128 x2
        acc = __builtin_amdgcn_wmma_f32_16x16x32_bf16(
            false, aa.b, false, bb.b, (short)0, acc, false, false);
      }
    }
  }

  const float bv = bias ? bias[n0 + ncol] : 0.0f;
#pragma unroll
  for (int v = 0; v < 8; ++v) {
    int gm = m0 + (half << 3) + v;        // D: VGPR v holds M=v (+8 hi half)
    out[gm * Cout + n0 + ncol] = acc[v] + bv;
  }
}

// ---------------------------------------------------------------------------
// Layout / precision conversion kernels
// ---------------------------------------------------------------------------
__global__ void nchw_to_nhwc_bf16(const float* __restrict__ src,
                                  unsigned short* __restrict__ dst,
                                  int Bn, int C, int H, int W, int n)
{
  int idx = blockIdx.x * blockDim.x + threadIdx.x;
  if (idx >= n) return;
  int w = idx % W; int t = idx / W;
  int h = t % H;   t /= H;
  int c = t % C;   int b = t / C;
  dst[((b * H + h) * W + w) * C + c] = f2bf(src[idx]);
}

// OIHW f32 -> per-lane WMMA B-fragment dwords:
// dst[((nt*K*K + tap)*NC + cc)*256 + lane*8 + v] = bf16 pair (k, k+1) for
// k = cc*32 + half*16 + v*2, co = nt*16 + (lane&15).
__global__ void pack_w_frag(const float* __restrict__ src,
                            unsigned int* __restrict__ dst,
                            int Co, int Ci, int K, int n)
{
  int idx = blockIdx.x * blockDim.x + threadIdx.x;
  if (idx >= n) return;
  int v    = idx & 7;
  int lane = (idx >> 3) & 31;
  int t    = idx >> 8;
  int NC   = Ci >> 5;
  int cc   = t % NC;       t /= NC;
  int tap  = t % (K * K);  t /= (K * K);
  int nt   = t;
  int half = lane >> 4, ncol = lane & 15;
  int ci   = cc * 32 + half * 16 + v * 2;
  int co   = nt * 16 + ncol;
  int kh   = tap / K, kw = tap % K;
  unsigned lo = f2bf(src[((co * Ci + ci)     * K + kh) * K + kw]);
  unsigned hi = f2bf(src[((co * Ci + ci + 1) * K + kh) * K + kw]);
  dst[idx] = lo | (hi << 16);
}

// ---------------------------------------------------------------------------
// Normalization statistics (LDS tree reductions)
// ---------------------------------------------------------------------------
__global__ void __launch_bounds__(256)
instance_stats(const float* __restrict__ y, float* __restrict__ mean,
               float* __restrict__ rstd, int HW, int C)
{
  int g = blockIdx.x;              // g = b*C + c
  int b = g / C, c = g % C;
  const float* p = y + (long)b * HW * C + c;
  float s = 0.f, s2 = 0.f;
  for (int i = threadIdx.x; i < HW; i += 256) {
    float v = p[(long)i * C];
    s += v; s2 += v * v;
  }
  __shared__ float sh1[256], sh2[256];
  sh1[threadIdx.x] = s; sh2[threadIdx.x] = s2;
  __syncthreads();
  for (int o = 128; o > 0; o >>= 1) {
    if (threadIdx.x < o) {
      sh1[threadIdx.x] += sh1[threadIdx.x + o];
      sh2[threadIdx.x] += sh2[threadIdx.x + o];
    }
    __syncthreads();
  }
  if (threadIdx.x == 0) {
    float m = sh1[0] / HW;
    float v = sh2[0] / HW - m * m;
    mean[g] = m;
    rstd[g] = rsqrtf(v + 1e-5f);
  }
}

__global__ void __launch_bounds__(256)
batch_stats(const float* __restrict__ y, float* __restrict__ mean,
            float* __restrict__ rstd, int rows, int C)
{
  int c = blockIdx.x;
  float s = 0.f, s2 = 0.f;
  for (int i = threadIdx.x; i < rows; i += 256) {
    float v = y[(long)i * C + c];
    s += v; s2 += v * v;
  }
  __shared__ float sh1[256], sh2[256];
  sh1[threadIdx.x] = s; sh2[threadIdx.x] = s2;
  __syncthreads();
  for (int o = 128; o > 0; o >>= 1) {
    if (threadIdx.x < o) {
      sh1[threadIdx.x] += sh1[threadIdx.x + o];
      sh2[threadIdx.x] += sh2[threadIdx.x + o];
    }
    __syncthreads();
  }
  if (threadIdx.x == 0) {
    float m = sh1[0] / rows;
    float v = sh2[0] / rows - m * m;
    mean[c] = m;
    rstd[c] = rsqrtf(v + 1e-5f);
  }
}

__global__ void in_norm_relu_bf16(const float* __restrict__ y,
                                  const float* __restrict__ mean,
                                  const float* __restrict__ rstd,
                                  unsigned short* __restrict__ dst,
                                  int HW, int C, int n)
{
  int idx = blockIdx.x * blockDim.x + threadIdx.x;
  if (idx >= n) return;
  int c = idx % C;
  int g = (idx / C / HW) * C + c;
  float v = (y[idx] - mean[g]) * rstd[g];
  dst[idx] = f2bf(fmaxf(v, 0.0f));
}

__global__ void in_norm_f32(float* __restrict__ y,
                            const float* __restrict__ mean,
                            const float* __restrict__ rstd,
                            int HW, int C, int n)
{
  int idx = blockIdx.x * blockDim.x + threadIdx.x;
  if (idx >= n) return;
  int c = idx % C;
  int g = (idx / C / HW) * C + c;
  y[idx] = (y[idx] - mean[g]) * rstd[g];
}

__global__ void bn_norm_relu_bf16(const float* __restrict__ y,
                                  const float* __restrict__ mean,
                                  const float* __restrict__ rstd,
                                  const float* __restrict__ gam,
                                  const float* __restrict__ bet,
                                  unsigned short* __restrict__ dst,
                                  int C, int n)
{
  int idx = blockIdx.x * blockDim.x + threadIdx.x;
  if (idx >= n) return;
  int c = idx % C;
  float v = (y[idx] - mean[c]) * rstd[c] * gam[c] + bet[c];
  dst[idx] = f2bf(fmaxf(v, 0.0f));
}

// ---------------------------------------------------------------------------
// SFOM: mean-of-DCT as a closed-form weighted reduction; idct(g*dct(x))==g*x.
// ---------------------------------------------------------------------------
__global__ void dct_mean_weights(float* __restrict__ wv, int N)
{
  int j = blockIdx.x * blockDim.x + threadIdx.x;
  if (j >= N) return;
  double th = 3.14159265358979323846 * (2.0 * j + 1.0) / (2.0 * N);
  double csum = sin(0.5 * N * th) * cos(0.5 * (N - 1) * th) / sin(0.5 * th);
  double s  = sqrt(2.0 / N);
  double s0 = 1.0 / sqrt((double)N);
  wv[j] = (float)((s * (csum - 1.0) + s0) / (double)N);   // fold the 1/N mean
}

__global__ void __launch_bounds__(256)
sfom_mean(const float* __restrict__ o, const float* __restrict__ wv,
          float* __restrict__ m, int HW, int C)
{
  int g = blockIdx.x;              // b*C + c
  int b = g / C, c = g % C;
  const float* p = o + (long)b * HW * C + c;
  float s = 0.f;
  for (int i = threadIdx.x; i < HW; i += 256) s += p[(long)i * C] * wv[i];
  __shared__ float sh[256];
  sh[threadIdx.x] = s;
  __syncthreads();
  for (int t = 128; t > 0; t >>= 1) {
    if (threadIdx.x < t) sh[threadIdx.x] += sh[threadIdx.x + t];
    __syncthreads();
  }
  if (threadIdx.x == 0) m[g] = sh[0];
}

__global__ void sfom_gate_k(const float* __restrict__ m,
                            const float* __restrict__ w1,   // [R,C]
                            const float* __restrict__ w2,   // [C,R]
                            float* __restrict__ gate, int C, int R)
{
  __shared__ float hid[16];
  for (int b = 0; b < 4; ++b) {
    if ((int)threadIdx.x < R) {
      float s = 0.f;
      for (int c = 0; c < C; ++c) s += m[b * C + c] * w1[threadIdx.x * C + c];
      hid[threadIdx.x] = fmaxf(s, 0.0f);
    }
    __syncthreads();
    if ((int)threadIdx.x < C) {
      float s = 0.f;
      for (int r = 0; r < R; ++r) s += hid[r] * w2[threadIdx.x * R + r];
      gate[b * C + threadIdx.x] = sigm(s);
    }
    __syncthreads();
  }
}

__global__ void sfom_apply_k(float* __restrict__ o,
                             const float* __restrict__ gate,
                             unsigned short* __restrict__ actbf,
                             int HW, int C, int n)
{
  int idx = blockIdx.x * blockDim.x + threadIdx.x;
  if (idx >= n) return;
  int c = idx % C;
  int g = (idx / C / HW) * C + c;
  float v  = o[idx];
  float r  = sigm(v * (1.0f + gate[g]) * 0.5f) * v;
  o[idx]     = r;
  actbf[idx] = f2bf(r);
}

// ---------------------------------------------------------------------------
// Tail: 1x1 sigmoid conv (32->1) and final NHWC->NCHW attention multiply
// ---------------------------------------------------------------------------
__global__ void sa_conv_k(const unsigned short* __restrict__ h,  // NHWC, C=32
                          const float* __restrict__ cw, const float* __restrict__ cb,
                          float* __restrict__ sa, int npix)
{
  int p = blockIdx.x * blockDim.x + threadIdx.x;
  if (p >= npix) return;
  float s = cb[0];
#pragma unroll
  for (int c = 0; c < 32; ++c) s += bf2f(h[p * 32 + c]) * cw[c];
  sa[p] = sigm(s);
}

__global__ void final_mul_k(const float* __restrict__ outf,   // NHWC
                            const float* __restrict__ sa,
                            float* __restrict__ dst,          // NCHW
                            int C, int H, int W, int n)
{
  int idx = blockIdx.x * blockDim.x + threadIdx.x;
  if (idx >= n) return;
  int w = idx % W; int t = idx / W;
  int h = t % H;   t /= H;
  int c = t % C;   int b = t / C;
  int p = (b * H + h) * W + w;
  dst[idx] = sa[p] * outf[(long)p * C + c];
}

// ---------------------------------------------------------------------------
// Host driver.  Scratch plan (needs ws_size >= ~119 MB):
//   [0, 4.6MB)  fragment-packed bf16 weights
//   [4.6, 5.1)  stats / gate / DCT weight / sa buffers
//   3 x 36MB slots with lifetime reuse (xb/a1/act | y1/conv | y2->out)
// ---------------------------------------------------------------------------
extern "C" void kernel_launch(void* const* d_in, const int* in_sizes, int n_in,
                              void* d_out, int out_size, void* d_ws, size_t ws_size,
                              hipStream_t stream)
{
  (void)in_sizes; (void)n_in; (void)out_size; (void)ws_size;

  const float* x    = (const float*)d_in[0];
  const float* w1   = (const float*)d_in[1];
  const float* b1   = (const float*)d_in[2];
  const float* w2   = (const float*)d_in[3];
  const float* b2   = (const float*)d_in[4];
  const float* saw1 = (const float*)d_in[5];
  const float* saw2 = (const float*)d_in[6];
  const float* dw[6]; const float* bg[6]; const float* bb[6];
  for (int i = 0; i < 6; ++i) {
    dw[i] = (const float*)d_in[7 + 3 * i];
    bg[i] = (const float*)d_in[8 + 3 * i];
    bb[i] = (const float*)d_in[9 + 3 * i];
  }
  const float* cw = (const float*)d_in[25];
  const float* cb = (const float*)d_in[26];

  const int Bn = 4, C = 128, H0 = 128, H1 = 130, H2 = 132, R = 8;
  const int HW1 = H1 * H1, HW2 = H2 * H2;
  const int NX = Bn * C * H0 * H0;
  const int N1 = Bn * HW1 * C;
  const int N2 = Bn * HW2 * C;

  char* ws = (char*)d_ws;
  const size_t OFF_W1T = 0;
  const size_t OFF_W2T = 294912;
  const size_t OFF_DWT = 589824;
  const int ci[6] = {128, 32, 64, 128, 128, 64};
  const int co[6] = {32, 64, 128, 128, 64, 32};
  size_t dwoff[6]; size_t o = OFF_DWT;
  for (int i = 0; i < 6; ++i) { dwoff[i] = o; o += (size_t)49 * ci[i] * co[i] * 2; }

  const size_t OFF_SMALL = 4603904;
  float* meanb = (float*)(ws + OFF_SMALL + 0);
  float* rstdb = (float*)(ws + OFF_SMALL + 2048);
  float* mvec  = (float*)(ws + OFF_SMALL + 4096);
  float* gate  = (float*)(ws + OFF_SMALL + 6144);
  float* bnm   = (float*)(ws + OFF_SMALL + 8192);
  float* bnr   = (float*)(ws + OFF_SMALL + 8704);
  float* wvec  = (float*)(ws + OFF_SMALL + 16384);
  float* sabuf = (float*)(ws + OFF_SMALL + 131072);

  const size_t SLOT = 37748736;                        // 36 MB
  char* slot0 = ws + 5242880;
  char* slot1 = slot0 + SLOT;
  char* slot2 = slot1 + SLOT;

  unsigned short* xb   = (unsigned short*)slot1;       // then a1, then SPEM act
  float*          y1   = (float*)slot2;                // then SPEM conv out
  float*          outb = (float*)slot0;                // y2 -> out (persists)
  unsigned int*   w1t  = (unsigned int*)(ws + OFF_W1T);
  unsigned int*   w2t  = (unsigned int*)(ws + OFF_W2T);

  const int TB = 256;
  auto blk = [](int n, int t) { return (n + t - 1) / t; };

  // ---- input / weight conversion (weights -> WMMA fragment layout) -------
  nchw_to_nhwc_bf16<<<blk(NX, TB), TB, 0, stream>>>(x, xb, Bn, C, H0, H0, NX);
  {
    int n = C * C * 9 / 2;    // dwords
    pack_w_frag<<<blk(n, TB), TB, 0, stream>>>(w1, w1t, C, C, 3, n);
    pack_w_frag<<<blk(n, TB), TB, 0, stream>>>(w2, w2t, C, C, 3, n);
  }
  for (int i = 0; i < 6; ++i) {
    int n = co[i] * ci[i] * 49 / 2;
    pack_w_frag<<<blk(n, TB), TB, 0, stream>>>(
        dw[i], (unsigned int*)(ws + dwoff[i]), co[i], ci[i], 7, n);
  }

  // ---- conv1 + instance-norm + relu --------------------------------------
  {
    int Mt = (Bn * HW1) / 16;
    dim3 g(blk(Mt, 8), C / 16);
    size_t sh = 9 * 4 * 1024;                          // taps * NC KB
    conv_wmma_kernel<3, 1, 128><<<g, 256, sh, stream>>>(
        xb, w1t, b1, y1, Bn, H0, H0, C, H1, H1);
  }
  instance_stats<<<Bn * C, 256, 0, stream>>>(y1, meanb, rstdb, HW1, C);
  unsigned short* a1 = (unsigned short*)slot1;
  in_norm_relu_bf16<<<blk(N1, TB), TB, 0, stream>>>(y1, meanb, rstdb, a1, HW1, C, N1);

  // ---- conv2 + instance-norm ---------------------------------------------
  {
    int Mt = (Bn * HW2) / 16;
    dim3 g(blk(Mt, 8), C / 16);
    size_t sh = 9 * 4 * 1024;
    conv_wmma_kernel<3, 1, 128><<<g, 256, sh, stream>>>(
        a1, w2t, b2, outb, Bn, H1, H1, C, H2, H2);
  }
  instance_stats<<<Bn * C, 256, 0, stream>>>(outb, meanb, rstdb, HW2, C);
  in_norm_f32<<<blk(N2, TB), TB, 0, stream>>>(outb, meanb, rstdb, HW2, C, N2);

  // ---- SFOM (DCT collapsed: idct(g*dct(x)) = g*x; mean via closed form) --
  dct_mean_weights<<<blk(HW2, TB), TB, 0, stream>>>(wvec, HW2);
  sfom_mean<<<Bn * C, 256, 0, stream>>>(outb, wvec, mvec, HW2, C);
  sfom_gate_k<<<1, 128, 0, stream>>>(mvec, saw1, saw2, gate, C, R);
  unsigned short* act = (unsigned short*)slot1;
  sfom_apply_k<<<blk(N2, TB), TB, 0, stream>>>(outb, gate, act, HW2, C, N2);

  // ---- SPEM dynamic-conv pyramid (7x7 WMMA conv + BN + relu) -------------
  float* convf = (float*)slot2;
  for (int i = 0; i < 6; ++i) {
    int Mt = (Bn * HW2) / 16;
    dim3 g(blk(Mt, 8), co[i] / 16);
    const unsigned int* wf = (const unsigned int*)(ws + dwoff[i]);
    size_t sh = (size_t)49 * (ci[i] / 32) * 1024;      // LDS weight stage
    switch (ci[i]) {
      case 128:
        conv_wmma_kernel<7, 0, 128><<<g, 256, sh, stream>>>(
            act, wf, nullptr, convf, Bn, H2, H2, co[i], H2, H2);
        break;
      case 64:
        conv_wmma_kernel<7, 0, 64><<<g, 256, sh, stream>>>(
            act, wf, nullptr, convf, Bn, H2, H2, co[i], H2, H2);
        break;
      default:
        conv_wmma_kernel<7, 0, 32><<<g, 256, sh, stream>>>(
            act, wf, nullptr, convf, Bn, H2, H2, co[i], H2, H2);
        break;
    }
    batch_stats<<<co[i], 256, 0, stream>>>(convf, bnm, bnr, Bn * HW2, co[i]);
    int n = Bn * HW2 * co[i];
    bn_norm_relu_bf16<<<blk(n, TB), TB, 0, stream>>>(convf, bnm, bnr, bg[i], bb[i],
                                                     act, co[i], n);
  }

  // ---- spatial attention + final multiply --------------------------------
  sa_conv_k<<<blk(Bn * HW2, TB), TB, 0, stream>>>(act, cw, cb, sabuf, Bn * HW2);
  final_mul_k<<<blk(N2, TB), TB, 0, stream>>>(outb, sabuf, (float*)d_out,
                                              C, H2, H2, N2);
}
